// Generator_78786880078112
// MI455X (gfx1250) — matrix operand order repaired and checked by reading, not compiled, not run on hardware
//
#include <hip/hip_runtime.h>
#include <hip/hip_bf16.h>

// ---------------- problem constants ----------------
constexpr int V_ = 32000, E_ = 256, H_ = 512, B_ = 16, S_ = 128;
constexpr int G4H = 4 * H_;  // 2048 gate columns
constexpr int NS = 4;        // timesteps per WG in the output kernel

typedef __bf16 bf16_t;
typedef __attribute__((ext_vector_type(16))) __bf16 v16bf;
typedef __attribute__((ext_vector_type(2)))  __bf16 v2bf;
typedef __attribute__((ext_vector_type(8)))  float  v8f;

// ---------------- workspace layout (bytes) ----------------
constexpr size_t OFF_EMBBF  = 0;                                        // bf16 [V][E]
constexpr size_t OFF_WOUTBF = OFF_EMBBF  + (size_t)V_ * E_ * 2;         // bf16 [V][H]
constexpr size_t OFF_WIH0BF = OFF_WOUTBF + (size_t)V_ * H_ * 2;         // bf16 [4H][E]
constexpr size_t OFF_WHH0BF = OFF_WIH0BF + (size_t)G4H * E_ * 2;        // bf16 [4H][H]
constexpr size_t OFF_WIH1BF = OFF_WHH0BF + (size_t)G4H * H_ * 2;        // bf16 [4H][H]
constexpr size_t OFF_WHH1BF = OFF_WIH1BF + (size_t)G4H * H_ * 2;        // bf16 [4H][H]
constexpr size_t OFF_B0SUM  = OFF_WHH1BF + (size_t)G4H * H_ * 2;        // f32 [4H]
constexpr size_t OFF_B1SUM  = OFF_B0SUM  + (size_t)G4H * 4;             // f32 [4H]
constexpr size_t OFF_G0X    = OFF_B1SUM  + (size_t)G4H * 4;             // f32 [S][B][4H]
constexpr size_t OFF_HSBF   = OFF_G0X    + (size_t)S_ * B_ * G4H * 4;   // bf16 [S][B][H]
constexpr size_t OFF_H0BF   = OFF_HSBF   + (size_t)S_ * B_ * H_ * 2;    // bf16 [2][B][H] dbl-buf
constexpr size_t OFF_H1BF   = OFF_H0BF   + (size_t)2 * B_ * H_ * 2;     // bf16 [2][B][H] dbl-buf
constexpr size_t OFF_C0ST   = OFF_H1BF   + (size_t)2 * B_ * H_ * 2;     // f32 [B][H]
constexpr size_t OFF_C1ST   = OFF_C0ST   + (size_t)B_ * H_ * 4;         // f32 [B][H]
constexpr size_t OFF_CNT    = OFF_C1ST   + (size_t)B_ * H_ * 4;         // u32 barrier counter
constexpr size_t WS_NEEDED  = OFF_CNT + 256;                            // ~73.5 MB

// ---------------- small device helpers ----------------
__device__ __forceinline__ v8f wmma_bf(v16bf a, v16bf b, v8f c) {
  return __builtin_amdgcn_wmma_f32_16x16x32_bf16(false, a, false, b, (short)0, c, false, false);
}

// A(16x32 bf16) lane pattern: lanes<16 hold K {0..15}, lanes>=16 hold {16..31}
__device__ __forceinline__ v16bf load_A_tile(const bf16_t* rowbase /* lane's row @ kbase */, int half) {
  v16bf a;
#pragma unroll
  for (int j = 0; j < 8; ++j) {
    int K = 2 * j + 16 * half;
    v2bf p = *reinterpret_cast<const v2bf*>(rowbase + K);
    a[2 * j] = p[0];
    a[2 * j + 1] = p[1];
  }
  return a;
}

__device__ __forceinline__ float sigm_(float x) { return 1.0f / (1.0f + __expf(-x)); }
__device__ __forceinline__ float tanh_(float x) {
  float ax = fabsf(x);
  float e  = __expf(-2.0f * ax);
  float r  = (1.0f - e) / (1.0f + e);
  return copysignf(r, x);
}

__device__ __forceinline__ void grid_barrier(unsigned* cnt, unsigned nwg, unsigned& phase) {
  __threadfence();
  __syncthreads();
  ++phase;
  if (threadIdx.x == 0) {
    atomicAdd(cnt, 1u);
    while (atomicAdd(cnt, 0u) < phase * nwg) { __builtin_amdgcn_s_sleep(1); }
  }
  __syncthreads();
  __threadfence();
}

// ---------------- kernel 1: prep (convert weights to bf16, fuse biases, init state) --------
__global__ void prep_kernel(const float* emb, const float* Wih0, const float* Whh0,
                            const float* bih0, const float* bhh0,
                            const float* Wih1, const float* Whh1,
                            const float* bih1, const float* bhh1,
                            const float* Wout, const float* h0, const float* c0,
                            bf16_t* embbf, bf16_t* woutbf, bf16_t* wih0bf,
                            bf16_t* whh0bf, bf16_t* wih1bf, bf16_t* whh1bf,
                            float* b0sum, float* b1sum,
                            bf16_t* h0bf, bf16_t* h1bf, float* c0st, float* c1st,
                            unsigned* cnt) {
  const size_t N_EMB = (size_t)V_ * E_;
  const size_t N_WO  = (size_t)V_ * H_;
  const size_t N_WI0 = (size_t)G4H * E_;
  const size_t N_WHH = (size_t)G4H * H_;
  const size_t N_B   = (size_t)G4H;
  const size_t N_H   = (size_t)B_ * H_;
  const size_t TOTAL = N_EMB + N_WO + N_WI0 + 3 * N_WHH + 2 * N_B + 4 * N_H;

  size_t idx    = (size_t)blockIdx.x * blockDim.x + threadIdx.x;
  size_t stride = (size_t)gridDim.x * blockDim.x;
  if (idx == 0) *cnt = 0u;

  for (size_t i = idx; i < TOTAL; i += stride) {
    size_t o = i;
    if (o < N_EMB)      { embbf[o]  = (bf16_t)emb[o];  continue; }  o -= N_EMB;
    if (o < N_WO)       { woutbf[o] = (bf16_t)Wout[o]; continue; }  o -= N_WO;
    if (o < N_WI0)      { wih0bf[o] = (bf16_t)Wih0[o]; continue; }  o -= N_WI0;
    if (o < N_WHH)      { whh0bf[o] = (bf16_t)Whh0[o]; continue; }  o -= N_WHH;
    if (o < N_WHH)      { wih1bf[o] = (bf16_t)Wih1[o]; continue; }  o -= N_WHH;
    if (o < N_WHH)      { whh1bf[o] = (bf16_t)Whh1[o]; continue; }  o -= N_WHH;
    if (o < N_B)        { b0sum[o]  = bih0[o] + bhh0[o]; continue; } o -= N_B;
    if (o < N_B)        { b1sum[o]  = bih1[o] + bhh1[o]; continue; } o -= N_B;
    if (o < N_H)        { h0bf[o]   = (bf16_t)h0[o];       continue; } o -= N_H;
    if (o < N_H)        { h1bf[o]   = (bf16_t)h0[N_H + o]; continue; } o -= N_H;
    if (o < N_H)        { c0st[o]   = c0[o];               continue; } o -= N_H;
    /* last region */     c1st[o]   = c0[N_H + o];
  }
}

// ---------------- kernel 2: G0x[s] = emb[y[:,s]] @ Wih0^T + (bih0+bhh0)  (parallel over s) --
__global__ void xproj_kernel(const int* y, const bf16_t* embbf, const bf16_t* wih0bf,
                             const float* b0sum, float* g0x) {
  int s = blockIdx.x;
  int tid = threadIdx.x, lane = tid & 31, w = tid >> 5;  // 4 waves / WG
  int n = lane & 15, half = lane >> 4, m = lane & 15;
  int rid = y[m * S_ + s];
  const bf16_t* arow = embbf + (size_t)rid * E_;

  for (int nt = w; nt < G4H / 16; nt += 4) {
    int nbase = nt * 16;
    float bb = b0sum[nbase + n];
    v8f c;
#pragma unroll
    for (int r = 0; r < 8; ++r) c[r] = bb;
#pragma unroll
    for (int kc = 0; kc < E_ / 32; ++kc) {
      v16bf a = load_A_tile(arow + kc * 32, half);
      v16bf b = *reinterpret_cast<const v16bf*>(wih0bf + (size_t)(nbase + n) * E_ + kc * 32 + 16 * half);
      c = wmma_bf(a, b, c);
    }
#pragma unroll
    for (int r = 0; r < 8; ++r) {
      int brow = 8 * half + r;
      g0x[((size_t)s * B_ + brow) * G4H + nbase + n] = c[r];
    }
  }
}

// ---------------- kernel 3: persistent LSTM recurrence (8 WGs x 4 waves = 32 j-tiles) ------
__global__ void lstm_kernel(const float* __restrict__ g0x,
                            const bf16_t* __restrict__ whh0bf,
                            const bf16_t* __restrict__ wih1bf,
                            const bf16_t* __restrict__ whh1bf,
                            const float* __restrict__ b1sum,
                            bf16_t* h0bf, bf16_t* h1bf,
                            float* c0st, float* c1st,
                            bf16_t* hsbf, unsigned* cnt) {
  int tid = threadIdx.x, lane = tid & 31;
  int gw = blockIdx.x * (blockDim.x >> 5) + (tid >> 5);  // 0..31: H-column tile (owns j, j+H, j+2H, j+3H)
  int jbase = gw * 16;
  int n = lane & 15, half = lane >> 4, mrow = lane & 15;
  unsigned phase = 0;

  for (int s = 0; s < S_; ++s) {
    const bf16_t* h0old = h0bf + (size_t)(s & 1) * (B_ * H_);
    bf16_t*       h0new = h0bf + (size_t)((s + 1) & 1) * (B_ * H_);
    const bf16_t* h1old = h1bf + (size_t)(s & 1) * (B_ * H_);
    bf16_t*       h1new = h1bf + (size_t)((s + 1) & 1) * (B_ * H_);

    // ---- layer 0: gates = G0x[s] (has x-proj + both biases) + h0_old @ Whh0^T
    v8f acc[4];
#pragma unroll
    for (int g = 0; g < 4; ++g) {
#pragma unroll
      for (int r = 0; r < 8; ++r)
        acc[g][r] = g0x[((size_t)s * B_ + 8 * half + r) * G4H + (size_t)g * H_ + jbase + n];
    }
    for (int kc = 0; kc < H_ / 32; ++kc) {
      v16bf a = load_A_tile(h0old + (size_t)mrow * H_ + kc * 32, half);
#pragma unroll
      for (int g = 0; g < 4; ++g) {
        v16bf b = *reinterpret_cast<const v16bf*>(whh0bf + (size_t)(g * H_ + jbase + n) * H_ + kc * 32 + 16 * half);
        acc[g] = wmma_bf(a, b, acc[g]);
      }
    }
#pragma unroll
    for (int r = 0; r < 8; ++r) {
      int brow = 8 * half + r, j = jbase + n;
      float ig = sigm_(acc[0][r]);
      float fg = sigm_(acc[1][r]);
      float gg = tanh_(acc[2][r]);
      float og = sigm_(acc[3][r]);
      float cold = c0st[brow * H_ + j];
      float cn = fg * cold + ig * gg;
      c0st[brow * H_ + j] = cn;
      h0new[brow * H_ + j] = (bf16_t)(og * tanh_(cn));
    }
    grid_barrier(cnt, 8, phase);

    // ---- layer 1: gates = h0_new @ Wih1^T + h1_old @ Whh1^T + (bih1+bhh1)
#pragma unroll
    for (int g = 0; g < 4; ++g) {
      float bb = b1sum[g * H_ + jbase + n];
#pragma unroll
      for (int r = 0; r < 8; ++r) acc[g][r] = bb;
    }
    for (int kc = 0; kc < H_ / 32; ++kc) {
      v16bf a = load_A_tile(h0new + (size_t)mrow * H_ + kc * 32, half);
#pragma unroll
      for (int g = 0; g < 4; ++g) {
        v16bf b = *reinterpret_cast<const v16bf*>(wih1bf + (size_t)(g * H_ + jbase + n) * H_ + kc * 32 + 16 * half);
        acc[g] = wmma_bf(a, b, acc[g]);
      }
    }
    for (int kc = 0; kc < H_ / 32; ++kc) {
      v16bf a = load_A_tile(h1old + (size_t)mrow * H_ + kc * 32, half);
#pragma unroll
      for (int g = 0; g < 4; ++g) {
        v16bf b = *reinterpret_cast<const v16bf*>(whh1bf + (size_t)(g * H_ + jbase + n) * H_ + kc * 32 + 16 * half);
        acc[g] = wmma_bf(a, b, acc[g]);
      }
    }
#pragma unroll
    for (int r = 0; r < 8; ++r) {
      int brow = 8 * half + r, j = jbase + n;
      float ig = sigm_(acc[0][r]);
      float fg = sigm_(acc[1][r]);
      float gg = tanh_(acc[2][r]);
      float og = sigm_(acc[3][r]);
      float cold = c1st[brow * H_ + j];
      float cn = fg * cold + ig * gg;
      c1st[brow * H_ + j] = cn;
      float hn = og * tanh_(cn);
      h1new[brow * H_ + j] = (bf16_t)hn;
      hsbf[((size_t)s * B_ + brow) * H_ + j] = (bf16_t)hn;
    }
    grid_barrier(cnt, 8, phase);
  }
}

// ---------------- kernel 4: logits = Hs @ Wout^T + bout, fused online softmax --------------
// 32 WGs x 256 thr (8 waves); each WG owns NS=4 timesteps (64 rows). d_out is logits scratch.
// LDS: 64 KB packed-A buffer; softmax reduction scratch is ALIASED on top of it (A is dead
// by the time reduction starts; an extra __syncthreads() guards the reuse).
__global__ void out_softmax_kernel(const bf16_t* __restrict__ hsbf,
                                   const bf16_t* __restrict__ woutbf,
                                   const float* __restrict__ bout, float* out) {
  __shared__ __align__(32) unsigned char smem[(size_t)NS * 16 * 32 * 16 * 2];  // exactly 64 KB
  bf16_t (*As)[16][32][16] = reinterpret_cast<bf16_t (*)[16][32][16]>(smem);
  float* redM = reinterpret_cast<float*>(smem);            // [8][2][NS][8]
  float* redL = redM + 8 * 2 * NS * 8;
  float* finM = redL + 8 * 2 * NS * 8;                     // [NS][16]
  float* finL = finM + NS * 16;

  int tid = threadIdx.x;
  int sbase = blockIdx.x * NS;

  // pack A tiles (NS timesteps) into WMMA lane order
  for (int slot = tid; slot < NS * 16 * 32; slot += 256) {
    int s = slot >> 9, kc = (slot >> 5) & 15, ln = slot & 31;
    int m = ln & 15, hf = ln >> 4;
    const bf16_t* row = hsbf + ((size_t)(sbase + s) * B_ + m) * H_ + kc * 32;
#pragma unroll
    for (int j = 0; j < 8; ++j) {
      int K = 2 * j + 16 * hf;
      v2bf p = *reinterpret_cast<const v2bf*>(row + K);
      As[s][kc][ln][2 * j] = p[0];
      As[s][kc][ln][2 * j + 1] = p[1];
    }
  }
  __syncthreads();

  int w = tid >> 5, lane = tid & 31, n = lane & 15, half = lane >> 4;
  float mr[NS][8], lr[NS][8];
#pragma unroll
  for (int s = 0; s < NS; ++s)
#pragma unroll
    for (int r = 0; r < 8; ++r) { mr[s][r] = -3.0e38f; lr[s][r] = 0.f; }

  for (int nt = w; nt < V_ / 16; nt += 8) {
    int vbase = nt * 16;
    float bb = bout[vbase + n];
    v8f cs[NS];
#pragma unroll
    for (int s = 0; s < NS; ++s)
#pragma unroll
      for (int r = 0; r < 8; ++r) cs[s][r] = bb;
#pragma unroll
    for (int kc = 0; kc < 16; ++kc) {
      v16bf b = *reinterpret_cast<const v16bf*>(woutbf + (size_t)(vbase + n) * H_ + kc * 32 + 16 * half);
#pragma unroll
      for (int s = 0; s < NS; ++s) {
        v16bf a = *reinterpret_cast<const v16bf*>(&As[s][kc][lane][0]);
        cs[s] = wmma_bf(a, b, cs[s]);
      }
    }
#pragma unroll
    for (int s = 0; s < NS; ++s) {
#pragma unroll
      for (int r = 0; r < 8; ++r) {
        int brow = 8 * half + r;
        float v = cs[s][r];
        out[((size_t)brow * S_ + sbase + s) * V_ + vbase + n] = v;
        if (v > mr[s][r]) { lr[s][r] *= __expf(mr[s][r] - v); mr[s][r] = v; }
        lr[s][r] += __expf(v - mr[s][r]);
      }
    }
  }

  // reduce across the 16 column-classes (xor network stays within each lane half)
#pragma unroll
  for (int s = 0; s < NS; ++s)
#pragma unroll
    for (int r = 0; r < 8; ++r)
#pragma unroll
      for (int off = 1; off < 16; off <<= 1) {
        float mo = __shfl_xor(mr[s][r], off, 32), lo = __shfl_xor(lr[s][r], off, 32);
        float M2 = fmaxf(mr[s][r], mo);
        lr[s][r] = lr[s][r] * __expf(mr[s][r] - M2) + lo * __expf(mo - M2);
        mr[s][r] = M2;
      }

  __syncthreads();  // everyone done reading As -> safe to alias reduction scratch over it
  if (n == 0) {
#pragma unroll
    for (int s = 0; s < NS; ++s)
#pragma unroll
      for (int r = 0; r < 8; ++r) {
        redM[(((w * 2 + half) * NS) + s) * 8 + r] = mr[s][r];
        redL[(((w * 2 + half) * NS) + s) * 8 + r] = lr[s][r];
      }
  }
  __syncthreads();
  if (tid < NS * 16) {
    int s = tid >> 4, b_ = tid & 15, hf = b_ >> 3, r = b_ & 7;
    float M = -3.0e38f, L = 0.f;
    for (int w2 = 0; w2 < 8; ++w2) {
      float mo = redM[(((w2 * 2 + hf) * NS) + s) * 8 + r];
      float lo = redL[(((w2 * 2 + hf) * NS) + s) * 8 + r];
      float M2 = fmaxf(M, mo);
      L = L * __expf(M - M2) + lo * __expf(mo - M2);
      M = M2;
    }
    finM[s * 16 + b_] = M;
    finL[s * 16 + b_] = L;
  }
  __syncthreads();

  // pass 2: normalize in place
  for (int s = 0; s < NS; ++s)
    for (int b_ = 0; b_ < 16; ++b_) {
      float M = finM[s * 16 + b_], Li = 1.0f / finL[s * 16 + b_];
      float* rowp = out + ((size_t)b_ * S_ + sbase + s) * V_;
      for (int v = tid; v < V_; v += 256) rowp[v] = __expf(rowp[v] - M) * Li;
    }
}

// ---------------- host glue ----------------
extern "C" void kernel_launch(void* const* d_in, const int* in_sizes, int n_in,
                              void* d_out, int out_size, void* d_ws, size_t ws_size,
                              hipStream_t stream) {
  (void)in_sizes; (void)n_in; (void)out_size; (void)ws_size;  // need ws_size >= WS_NEEDED (~74 MB)

  const int*   y    = (const int*)d_in[0];
  const float* emb  = (const float*)d_in[1];
  const float* Wih0 = (const float*)d_in[2];
  const float* Whh0 = (const float*)d_in[3];
  const float* bih0 = (const float*)d_in[4];
  const float* bhh0 = (const float*)d_in[5];
  const float* Wih1 = (const float*)d_in[6];
  const float* Whh1 = (const float*)d_in[7];
  const float* bih1 = (const float*)d_in[8];
  const float* bhh1 = (const float*)d_in[9];
  const float* Wout = (const float*)d_in[10];
  const float* bout = (const float*)d_in[11];
  const float* h0   = (const float*)d_in[12];
  const float* c0   = (const float*)d_in[13];

  char* ws = (char*)d_ws;
  bf16_t*   embbf  = (bf16_t*)(ws + OFF_EMBBF);
  bf16_t*   woutbf = (bf16_t*)(ws + OFF_WOUTBF);
  bf16_t*   wih0bf = (bf16_t*)(ws + OFF_WIH0BF);
  bf16_t*   whh0bf = (bf16_t*)(ws + OFF_WHH0BF);
  bf16_t*   wih1bf = (bf16_t*)(ws + OFF_WIH1BF);
  bf16_t*   whh1bf = (bf16_t*)(ws + OFF_WHH1BF);
  float*    b0sum  = (float*)(ws + OFF_B0SUM);
  float*    b1sum  = (float*)(ws + OFF_B1SUM);
  float*    g0x    = (float*)(ws + OFF_G0X);
  bf16_t*   hsbf   = (bf16_t*)(ws + OFF_HSBF);
  bf16_t*   h0bf   = (bf16_t*)(ws + OFF_H0BF);
  bf16_t*   h1bf   = (bf16_t*)(ws + OFF_H1BF);
  float*    c0st   = (float*)(ws + OFF_C0ST);
  float*    c1st   = (float*)(ws + OFF_C1ST);
  unsigned* cnt    = (unsigned*)(ws + OFF_CNT);

  prep_kernel<<<4096, 256, 0, stream>>>(emb, Wih0, Whh0, bih0, bhh0, Wih1, Whh1, bih1, bhh1,
                                        Wout, h0, c0, embbf, woutbf, wih0bf, whh0bf, wih1bf,
                                        whh1bf, b0sum, b1sum, h0bf, h1bf, c0st, c1st, cnt);
  xproj_kernel<<<S_, 128, 0, stream>>>(y, embbf, wih0bf, b0sum, g0x);
  lstm_kernel<<<8, 128, 0, stream>>>(g0x, whh0bf, wih1bf, whh1bf, b1sum,
                                     h0bf, h1bf, c0st, c1st, hsbf, cnt);
  out_softmax_kernel<<<S_ / NS, 256, 0, stream>>>(hsbf, woutbf, bout, (float*)d_out);
}
